// Decoder_2808908612305
// MI455X (gfx1250) — matrix operand order repaired and checked by reading, not compiled, not run on hardware
//
#include <hip/hip_runtime.h>
#include <math.h>

#define B_  128
#define T_  24
#define V_  12000
#define FD_ 2048
#define H_  512
#define E_  512
#define NV_ 49
#define XW_ (E_ + H_)   // 1024 = concat(emb, vg) width

typedef __attribute__((ext_vector_type(16))) _Float16 v16h;
typedef __attribute__((ext_vector_type(8)))  _Float16 v8h;
typedef __attribute__((ext_vector_type(8)))  float    v8f;
typedef _Float16 half_t;

__device__ __forceinline__ float sigmoidf_(float x) { return 1.0f / (1.0f + expf(-x)); }

// ---------------------------------------------------------------------------
// Dual-source f16 WMMA GEMM:  C[M,N] = act(A1*W1^T + A2*W2^T + b1 + b2)
// A: f16 row-major M x K (row stride lda), W: f16 row-major N x K (stride K).
// 8 waves/block tile 64(M) x 64(N); each wave: one A frag feeds two 16x16
// N-tiles (two v_wmma issues per 32-K chunk). All K multiples of 32; N ragged.
// Ragged-N handling: W row index is CLAMPED to N-1 (valid memory, garbage
// values only pollute output columns n>=N which are never stored — WMMA
// column n of D depends only on column n of B). M multiple of 16.
// Fragment layouts used here (contiguous k order, same in A and B):
//   A 16x32 f16: lane m=lane&15, kbA=(lane>>4)*16 -> 16 contiguous halves
//   B 32x16 f16 (=W^T): lane n=lane&15, kbB=(lane>>4)*16 -> 16 contiguous halves
// ---------------------------------------------------------------------------
__global__ __launch_bounds__(256) void gemm_dual_wmma(
    const half_t* __restrict__ A1, int lda1, const half_t* __restrict__ W1, int K1,
    const half_t* __restrict__ A2, int lda2, const half_t* __restrict__ W2, int K2,
    const float* __restrict__ bias1, const float* __restrict__ bias2,
    float* __restrict__ Cf, half_t* __restrict__ Ch, int ldc, int M, int N, int relu)
{
    const int lane = threadIdx.x & 31;
    const int wv   = threadIdx.x >> 5;
    const int m0 = blockIdx.y * 64 + (wv & 3) * 16;
    const int n0 = blockIdx.x * 64 + (wv >> 2) * 32;   // wave covers n0 and n0+16
    if (m0 >= M) return;                               // wave-uniform

    const int rA  = m0 + (lane & 15);
    const int rW0 = min(n0 + (lane & 15), N - 1);      // clamped (see header note)
    const int rW1 = min(n0 + 16 + (lane & 15), N - 1);
    const int kbA = (lane >> 4) * 16;
    const int kbB = (lane >> 4) * 16;

    v8f acc0 = {}, acc1 = {};

    {   // ---- term 1
        const half_t* pa  = A1 + (size_t)rA  * lda1;
        const half_t* pw0 = W1 + (size_t)rW0 * K1;
        const half_t* pw1 = W1 + (size_t)rW1 * K1;
        for (int k = 0; k < K1; k += 32) {
            v8h alo = *(const v8h*)(pa + k + kbA);
            v8h ahi = *(const v8h*)(pa + k + kbA + 8);
            v16h a = __builtin_shufflevector(alo, ahi,
                         0,1,2,3,4,5,6,7,8,9,10,11,12,13,14,15);
            v16h b0 = *(const v16h*)(pw0 + k + kbB);
            v16h b1 = *(const v16h*)(pw1 + k + kbB);
            acc0 = __builtin_amdgcn_wmma_f32_16x16x32_f16(false, a, false, b0,
                                                          (short)0, acc0, false, false);
            acc1 = __builtin_amdgcn_wmma_f32_16x16x32_f16(false, a, false, b1,
                                                          (short)0, acc1, false, false);
        }
    }
    if (A2) {  // ---- term 2 (accumulates)
        const half_t* pa  = A2 + (size_t)rA  * lda2;
        const half_t* pw0 = W2 + (size_t)rW0 * K2;
        const half_t* pw1 = W2 + (size_t)rW1 * K2;
        for (int k = 0; k < K2; k += 32) {
            v8h alo = *(const v8h*)(pa + k + kbA);
            v8h ahi = *(const v8h*)(pa + k + kbA + 8);
            v16h a = __builtin_shufflevector(alo, ahi,
                         0,1,2,3,4,5,6,7,8,9,10,11,12,13,14,15);
            v16h b0 = *(const v16h*)(pw0 + k + kbB);
            v16h b1 = *(const v16h*)(pw1 + k + kbB);
            acc0 = __builtin_amdgcn_wmma_f32_16x16x32_f16(false, a, false, b0,
                                                          (short)0, acc0, false, false);
            acc1 = __builtin_amdgcn_wmma_f32_16x16x32_f16(false, a, false, b1,
                                                          (short)0, acc1, false, false);
        }
    }

    // ---- epilogue: lane owns column n (and n+16), rows mb..mb+7
    const int mb = m0 + (lane >> 4) * 8;
    #pragma unroll
    for (int p = 0; p < 2; ++p) {
        const int n = n0 + (lane & 15) + p * 16;
        if (n < N) {
            float bv = 0.0f;
            if (bias1) bv += bias1[n];
            if (bias2) bv += bias2[n];
            const v8f& acc = p ? acc1 : acc0;
            #pragma unroll
            for (int r = 0; r < 8; ++r) {
                float v = acc[r] + bv;
                if (relu) v = fmaxf(v, 0.0f);
                if (Cf) Cf[(size_t)(mb + r) * ldc + n] = v;
                if (Ch) Ch[(size_t)(mb + r) * ldc + n] = (half_t)v;
            }
        }
    }
}

// ---------------------------------------------------------------------------
// Elementwise / small kernels
// ---------------------------------------------------------------------------
__global__ void cvt_f16_kernel(const float* __restrict__ src, half_t* __restrict__ dst, int n)
{
    int i = blockIdx.x * blockDim.x + threadIdx.x;
    if (i < n) dst[i] = (half_t)src[i];
}

__global__ void mean_feat_kernel(const float* __restrict__ f, half_t* __restrict__ mean)
{
    int idx = blockIdx.x * blockDim.x + threadIdx.x;     // B*FD
    if (idx >= B_ * FD_) return;
    int b = idx / FD_, d = idx % FD_;
    const float* p = f + (size_t)b * NV_ * FD_ + d;
    float s = 0.0f;
    for (int v = 0; v < NV_; ++v) s += p[(size_t)v * FD_];
    mean[idx] = (half_t)(s * (1.0f / (float)NV_));
}

__global__ void init_state_kernel(const float* __restrict__ vg,
                                  half_t* __restrict__ hs16, float* __restrict__ cs)
{
    int idx = blockIdx.x * blockDim.x + threadIdx.x;     // B*H
    if (idx >= B_ * H_) return;
    hs16[idx] = (half_t)vg[idx];
    cs[idx]   = vg[idx];
}

// X16[b][t][0:E] = embed_w[captions[b,t]] ; X16[b][t][E:E+H] = vg[b]
__global__ void build_x_kernel(const int* __restrict__ captions,
                               const float* __restrict__ embed_w,
                               const float* __restrict__ vg, half_t* __restrict__ X)
{
    int idx = blockIdx.x * blockDim.x + threadIdx.x;     // B*T*XW
    if (idx >= B_ * T_ * XW_) return;
    int c  = idx % XW_;
    int bt = idx / XW_;
    int b  = bt / T_;
    float v;
    if (c < E_) {
        int tok = captions[bt];
        v = embed_w[(size_t)tok * E_ + c];
    } else {
        v = vg[b * H_ + (c - E_)];
    }
    X[idx] = (half_t)v;
}

__global__ void lstm_cell_kernel(const float* __restrict__ gpre,
                                 const float* __restrict__ gates,
                                 float* __restrict__ cs, half_t* __restrict__ hs16,
                                 half_t* __restrict__ s16, half_t* __restrict__ ctxh, int t)
{
    int idx = blockIdx.x * blockDim.x + threadIdx.x;     // B*H
    if (idx >= B_ * H_) return;
    int b = idx / H_, h = idx % H_;
    const float* g = gates + (size_t)b * 4 * H_;
    float ig = sigmoidf_(g[h]);
    float fg = sigmoidf_(g[H_ + h]);
    float gg = tanhf(g[2 * H_ + h]);
    float og = sigmoidf_(g[3 * H_ + h]);
    float cn = fg * cs[idx] + ig * gg;
    float hn = og * tanhf(cn);
    cs[idx]   = cn;
    hs16[idx] = (half_t)hn;
    s16[idx]  = (half_t)(sigmoidf_(gpre[idx]) * cn);     // s = gate * c_new
    ctxh[((size_t)b * T_ + t) * (2 * H_) + H_ + h] = (half_t)hn; // right half = hidden
}

// per-batch attention scores: z, i, softmax -> a[b,49], beta[b]
__global__ void att_z_kernel(const float* __restrict__ fpre, const float* __restrict__ hatt,
                             const float* __restrict__ spre, const float* __restrict__ ctx_w,
                             const float* __restrict__ ctx_b,
                             float* __restrict__ a, float* __restrict__ beta)
{
    __shared__ float hsh[NV_], cw[NV_], zsh[NV_];
    int b = blockIdx.x;
    int v = threadIdx.x;
    if (v < NV_) { hsh[v] = hatt[b * NV_ + v]; cw[v] = ctx_w[v]; }
    __syncthreads();
    if (v < NV_) {
        const float* fr = fpre + ((size_t)b * NV_ + v) * NV_;
        float acc = 0.0f;
        for (int n = 0; n < NV_; ++n) acc += tanhf(fr[n] + hsh[n]) * cw[n];
        zsh[v] = acc + ctx_b[0];
    }
    __syncthreads();
    if (v == 0) {
        float iacc = 0.0f;
        for (int n = 0; n < NV_; ++n) iacc += tanhf(spre[b * NV_ + n] + hsh[n]) * cw[n];
        float iv = iacc + ctx_b[0];
        float m = iv;
        for (int j = 0; j < NV_; ++j) m = fmaxf(m, zsh[j]);
        float S = 0.0f;
        for (int j = 0; j < NV_; ++j) { float e = expf(zsh[j] - m); zsh[j] = e; S += e; }
        float ei  = expf(iv - m);
        beta[b]   = ei / (S + ei);                 // last slot of concat-softmax
        float inv = 1.0f / S;                      // a = softmax(z) (shift-invariant)
        for (int j = 0; j < NV_; ++j) a[b * NV_ + j] = zsh[j] * inv;
    }
}

// ctx = beta*s + (1-beta) * (a @ ft[b])  -> left half of ctxh (f16)
__global__ void att_ctx_kernel(const float* __restrict__ a, const float* __restrict__ beta,
                               const half_t* __restrict__ s16, const half_t* __restrict__ ft,
                               half_t* __restrict__ ctxh, int t)
{
    int idx = blockIdx.x * blockDim.x + threadIdx.x;     // B*H
    if (idx >= B_ * H_) return;
    int b = idx / H_, h = idx % H_;
    const half_t* fb = ft + (size_t)b * NV_ * H_ + h;
    const float*  ab = a + b * NV_;
    float acc = 0.0f;
    for (int v = 0; v < NV_; ++v) acc += ab[v] * (float)fb[(size_t)v * H_];
    float bt = beta[b];
    ctxh[((size_t)b * T_ + t) * (2 * H_) + h] =
        (half_t)(bt * (float)s16[idx] + (1.0f - bt) * acc);
}

// in-place log-softmax over rows of 12000 in d_out
__global__ __launch_bounds__(256) void log_softmax_kernel(float* __restrict__ out)
{
    __shared__ float red[256];
    const size_t row = blockIdx.x;
    float* p = out + row * (size_t)V_;
    int tid = threadIdx.x;
    float m = -3.4e38f;
    for (int j = tid; j < V_; j += 256) m = fmaxf(m, p[j]);
    red[tid] = m; __syncthreads();
    for (int s = 128; s > 0; s >>= 1) { if (tid < s) red[tid] = fmaxf(red[tid], red[tid + s]); __syncthreads(); }
    m = red[0]; __syncthreads();
    float ssum = 0.0f;
    for (int j = tid; j < V_; j += 256) ssum += expf(p[j] - m);
    red[tid] = ssum; __syncthreads();
    for (int s = 128; s > 0; s >>= 1) { if (tid < s) red[tid] += red[tid + s]; __syncthreads(); }
    float lse = m + logf(red[0]);
    for (int j = tid; j < V_; j += 256) p[j] -= lse;
}

// ---------------------------------------------------------------------------
extern "C" void kernel_launch(void* const* d_in, const int* in_sizes, int n_in,
                              void* d_out, int out_size, void* d_ws, size_t ws_size,
                              hipStream_t stream)
{
    (void)in_sizes; (void)n_in; (void)out_size; (void)ws_size;
    const float* features  = (const float*)d_in[0];
    const int*   captions  = (const int*)  d_in[1];
    const float* embed_w   = (const float*)d_in[2];
    const float* globalf_w = (const float*)d_in[3];
    const float* globalf_b = (const float*)d_in[4];
    const float* gate_x_w  = (const float*)d_in[5];
    const float* gate_x_b  = (const float*)d_in[6];
    const float* gate_h_w  = (const float*)d_in[7];
    const float* gate_h_b  = (const float*)d_in[8];
    const float* lstm_wih  = (const float*)d_in[9];
    const float* lstm_bih  = (const float*)d_in[10];
    const float* lstm_whh  = (const float*)d_in[11];
    const float* lstm_bhh  = (const float*)d_in[12];
    const float* fc_out_w  = (const float*)d_in[13];
    const float* fc_out_b  = (const float*)d_in[14];
    const float* att_ft_w  = (const float*)d_in[15];
    const float* att_ft_b  = (const float*)d_in[16];
    const float* att_f_w   = (const float*)d_in[17];
    const float* att_f_b   = (const float*)d_in[18];
    const float* att_h_w   = (const float*)d_in[19];
    const float* att_h_b   = (const float*)d_in[20];
    const float* att_ctx_w = (const float*)d_in[21];
    const float* att_ctx_b = (const float*)d_in[22];
    const float* att_s_w   = (const float*)d_in[23];
    const float* att_s_b   = (const float*)d_in[24];
    float* out = (float*)d_out;

    // --- workspace carve-up (64B-aligned blocks) ---
    char* base = (char*)d_ws;
    size_t off = 0;
    auto allocB = [&](size_t bytes) -> void* {
        void* p = base + off; off += (bytes + 63) & ~(size_t)63; return p;
    };
    auto allocF = [&](size_t n) -> float*  { return (float*) allocB(n * sizeof(float));  };
    auto allocH = [&](size_t n) -> half_t* { return (half_t*)allocB(n * sizeof(half_t)); };

    // f16 weight copies (converted once per launch; reused across 24 steps / 48 M-tiles)
    half_t* feat16   = allocH((size_t)B_ * NV_ * FD_);
    half_t* globalf16= allocH((size_t)H_ * FD_);
    half_t* gatex16  = allocH((size_t)H_ * XW_);
    half_t* gateh16  = allocH((size_t)H_ * H_);
    half_t* wih16    = allocH((size_t)4 * H_ * XW_);
    half_t* whh16    = allocH((size_t)4 * H_ * H_);
    half_t* fc16     = allocH((size_t)V_ * 2 * H_);
    half_t* attft16  = allocH((size_t)H_ * FD_);
    half_t* attf16   = allocH((size_t)NV_ * H_);
    half_t* atth16   = allocH((size_t)NV_ * H_);
    half_t* atts16   = allocH((size_t)NV_ * H_);
    // f16 activations
    half_t* mean16   = allocH((size_t)B_ * FD_);
    half_t* X16      = allocH((size_t)B_ * T_ * XW_);
    half_t* ft16     = allocH((size_t)B_ * NV_ * H_);
    half_t* hs16     = allocH((size_t)B_ * H_);
    half_t* s16      = allocH((size_t)B_ * H_);
    half_t* ctxh16   = allocH((size_t)B_ * T_ * 2 * H_);
    // f32 state / intermediates
    float* vg    = allocF((size_t)B_ * H_);
    float* fpre  = allocF((size_t)B_ * NV_ * NV_);
    float* cs    = allocF((size_t)B_ * H_);
    float* gpre  = allocF((size_t)B_ * H_);
    float* gates = allocF((size_t)B_ * 4 * H_);
    float* hatt  = allocF((size_t)B_ * NV_);
    float* spre  = allocF((size_t)B_ * NV_);
    float* abuf  = allocF((size_t)B_ * NV_);
    float* beta  = allocF((size_t)B_);

    auto cvt = [&](const float* src, half_t* dst, size_t n) {
        cvt_f16_kernel<<<(unsigned)((n + 255) / 256), 256, 0, stream>>>(src, dst, (int)n);
    };
    auto gemm = [&](const half_t* A1, int lda1, const half_t* W1, int K1,
                    const half_t* A2, int lda2, const half_t* W2, int K2,
                    const float* b1, const float* b2,
                    float* Cf, half_t* Ch, int ldc, int M, int N, int relu) {
        dim3 g((N + 63) / 64, (M + 63) / 64);
        gemm_dual_wmma<<<g, 256, 0, stream>>>(A1, lda1, W1, K1, A2, lda2, W2, K2,
                                              b1, b2, Cf, Ch, ldc, M, N, relu);
    };

    // ---- one-time f32 -> f16 operand conversion ----
    cvt(features,  feat16,   (size_t)B_ * NV_ * FD_);
    cvt(globalf_w, globalf16,(size_t)H_ * FD_);
    cvt(gate_x_w,  gatex16,  (size_t)H_ * XW_);
    cvt(gate_h_w,  gateh16,  (size_t)H_ * H_);
    cvt(lstm_wih,  wih16,    (size_t)4 * H_ * XW_);
    cvt(lstm_whh,  whh16,    (size_t)4 * H_ * H_);
    cvt(fc_out_w,  fc16,     (size_t)V_ * 2 * H_);
    cvt(att_ft_w,  attft16,  (size_t)H_ * FD_);
    cvt(att_f_w,   attf16,   (size_t)NV_ * H_);
    cvt(att_h_w,   atth16,   (size_t)NV_ * H_);
    cvt(att_s_w,   atts16,   (size_t)NV_ * H_);

    // ---- time-invariant precompute ----
    mean_feat_kernel<<<(B_ * FD_ + 255) / 256, 256, 0, stream>>>(features, mean16);
    gemm(mean16, FD_, globalf16, FD_, nullptr, 0, nullptr, 0,
         globalf_b, nullptr, vg, nullptr, H_, B_, H_, 1);                 // vg = relu(...)
    init_state_kernel<<<(B_ * H_ + 255) / 256, 256, 0, stream>>>(vg, hs16, cs);
    build_x_kernel<<<(B_ * T_ * XW_ + 255) / 256, 256, 0, stream>>>(captions, embed_w, vg, X16);
    gemm(feat16, FD_, attft16, FD_, nullptr, 0, nullptr, 0,
         att_ft_b, nullptr, nullptr, ft16, H_, B_ * NV_, H_, 1);          // ft = relu(...)
    gemm(ft16, H_, attf16, H_, nullptr, 0, nullptr, 0,
         att_f_b, nullptr, fpre, nullptr, NV_, B_ * NV_, NV_, 0);         // fpre

    // ---- sequential recurrence ----
    for (int t = 0; t < T_; ++t) {
        const half_t* Xt = X16 + (size_t)t * XW_;   // row stride T*XW
        gemm(Xt, T_ * XW_, gatex16, XW_, hs16, H_, gateh16, H_,
             gate_x_b, gate_h_b, gpre, nullptr, H_, B_, H_, 0);
        gemm(Xt, T_ * XW_, wih16, XW_, hs16, H_, whh16, H_,
             lstm_bih, lstm_bhh, gates, nullptr, 4 * H_, B_, 4 * H_, 0);
        lstm_cell_kernel<<<(B_ * H_ + 255) / 256, 256, 0, stream>>>(gpre, gates, cs, hs16, s16, ctxh16, t);
        gemm(hs16, H_, atth16, H_, nullptr, 0, nullptr, 0,
             att_h_b, nullptr, hatt, nullptr, NV_, B_, NV_, 0);
        gemm(s16, H_, atts16, H_, nullptr, 0, nullptr, 0,
             att_s_b, nullptr, spre, nullptr, NV_, B_, NV_, 0);
        att_z_kernel<<<B_, 64, 0, stream>>>(fpre, hatt, spre, att_ctx_w, att_ctx_b, abuf, beta);
        att_ctx_kernel<<<(B_ * H_ + 255) / 256, 256, 0, stream>>>(abuf, beta, s16, ft16, ctxh16, t);
    }

    // ---- deferred output projection (60% of FLOPs) + in-place log-softmax ----
    gemm(ctxh16, 2 * H_, fc16, 2 * H_, nullptr, 0, nullptr, 0,
         fc_out_b, nullptr, out, nullptr, V_, B_ * T_, V_, 0);
    log_softmax_kernel<<<B_ * T_, 256, 0, stream>>>(out);
}
